// PointNet2V3Classifier_42812234007146
// MI455X (gfx1250) — matrix operand order repaired and checked by reading, not compiled, not run on hardware
//
#include <hip/hip_runtime.h>
#include <hip/hip_bf16.h>
#include <math.h>

// ---------------------------------------------------------------------------
// PointNet++ classifier for MI455X (gfx1250, wave32, WMMA).
//
// ~25-30 GFLOP, working set ~28MB << 192MB L2 -> compute bound. All channel-
// mixing GEMMs run on v_wmma_f32_16x16x32_f16 (f32 accumulate). Conv stacks
// are fused per workgroup in LDS (gather -> GEMM1 -> GEMM2 -> GEMM3 -> max).
//
// Data path tuned for the WMMA pipe:
//  * weights pre-converted to f16 and pre-swizzled into A-fragment order
//    (one 512-half block per 16x32 tile; each lane's 16 halves contiguous)
//    -> A fragment = 2x global_load_b128 from L2, no cvt in the hot loop.
//  * activations stored [col][chan] in LDS -> B fragment = 2x ds_load_b128,
//    C writeback = 1x packed ds_store_b128 per lane.
// ---------------------------------------------------------------------------

#define BATCH 16
#define NPTS  32768
#define NCLS  12

typedef _Float16 half_t;
typedef __attribute__((ext_vector_type(16))) _Float16 v16h;
typedef __attribute__((ext_vector_type(8)))  _Float16 v8h;
typedef __attribute__((ext_vector_type(8)))  float    v8f;

// ---------------------------------------------------------------------------
// WMMA helpers (layouts per CDNA5 ISA, wave32)
// ---------------------------------------------------------------------------
__device__ inline v8f wmma32(v16h a, v16h b, v8f c) {
  return __builtin_amdgcn_wmma_f32_16x16x32_f16(
      /*neg_a=*/false, a, /*neg_b=*/false, b,
      /*c_mod=*/(short)0, c, /*reuse_a=*/false, /*reuse_b=*/false);
}

__device__ inline v16h cat8(v8h lo, v8h hi) {
  return __builtin_shufflevector(lo, hi, 0, 1, 2, 3, 4, 5, 6, 7,
                                 8, 9, 10, 11, 12, 13, 14, 15);
}

// A fragment from pre-swizzled f16 weights: fragment (mt,kt) occupies 512
// contiguous halves; lane's 16 halves are 32 contiguous bytes.
__device__ inline v16h load_A_sw(const half_t* __restrict__ Wsw, int kfrags,
                                 int mt, int kt, int lane) {
  const half_t* p = Wsw + (((size_t)(mt * kfrags + kt)) << 9) + (lane << 4);
  v8h lo = *(const v8h*)p;
  v8h hi = *(const v8h*)(p + 8);
  return cat8(lo, hi);
}

// B fragment (32x16) from LDS activations stored [col][chan] (pitch mult. 8):
// lane<16 holds K 0..15, lane>=16 holds K 16..31; n = lane&15.
__device__ inline v16h load_B_lds(const half_t* Bm, int pitch,
                                  int kbase, int nbase, int lane) {
  const half_t* p = Bm + (size_t)(nbase + (lane & 15)) * pitch + kbase +
                    ((lane & 16) ? 16 : 0);
  v8h lo = *(const v8h*)p;
  v8h hi = *(const v8h*)(p + 8);
  return cat8(lo, hi);
}

// B fragment from global f32 matrix F[n][k] (row pitch ldk); n = batch lane.
__device__ inline v16h load_B_gf32(const float* F, int ldk, int kmax,
                                   int kbase, int lane) {
  v16h b;
  int n  = lane & 15;
  int kb = kbase + ((lane & 16) ? 16 : 0);
#pragma unroll
  for (int h = 0; h < 16; ++h) {
    int k = kb + h;
    b[h] = (half_t)((k < kmax) ? F[(size_t)n * ldk + k] : 0.f);
  }
  return b;
}

// Fold conv-bias + batchnorm into per-channel scale/shift:
//   y = conv*sc + sh,  sc = g*rsqrt(rv+eps),  sh = (bias-rm)*sc + b
__device__ inline void stage_scsh(float2* dst, const float* g, const float* bb,
                                  const float* rm, const float* rv,
                                  const float* bias, int C, int tid) {
  for (int i = tid; i < C; i += 256) {
    float sc = g[i] * rsqrtf(rv[i] + 1e-5f);
    dst[i] = make_float2(sc, (bias[i] - rm[i]) * sc + bb[i]);
  }
}

// One GEMM stage: D(f16,LDS,[col][chan]) = act( scsh( A_sw @ B ) )
template <bool RELU>
__device__ inline void gemm_stage(const half_t* __restrict__ Wsw, int M,
                                  int kfrags, const half_t* Bsrc, int pitchB,
                                  half_t* Dst, int pitchD, int ncol,
                                  const float2* ss, int wave, int lane) {
  const int mt_n  = M / 16;
  const int tiles = mt_n * (ncol / 16);
  for (int t = wave; t < tiles; t += 8) {
    int mt = t % mt_n, nt = t / mt_n;
    v8f c = {};
    for (int kt = 0; kt < kfrags; ++kt) {
      v16h a = load_A_sw(Wsw, kfrags, mt, kt, lane);
      v16h b = load_B_lds(Bsrc, pitchB, kt * 32, nt * 16, lane);
      c = wmma32(a, b, c);
    }
    int n  = nt * 16 + (lane & 15);
    int m0 = mt * 16 + ((lane & 16) ? 8 : 0);
    v8h pk;
#pragma unroll
    for (int r = 0; r < 8; ++r) {
      float y = c[r] * ss[m0 + r].x + ss[m0 + r].y;
      if (RELU) y = fmaxf(y, 0.f);
      pk[r] = (half_t)y;
    }
    *(v8h*)&Dst[(size_t)n * pitchD + m0] = pk;
  }
}

// Final GEMM + per-centroid max over the K (neighbor) axis. Each task owns
// (Mtile, centroid), loops the centroid's N-tiles so the reduction stays in
// registers; cross-lane max via xor-shuffles inside each 16-lane half
// (halves carry different M rows, so offsets 8..1 only).
__device__ inline void gemm3_max(const half_t* __restrict__ Wsw, int M,
                                 int kfrags, const half_t* Bsrc, int pitchB,
                                 const float2* ss, int knb, int ncol,
                                 float* outb, int S, int sbase,
                                 int wave, int lane) {
  const int mt_n   = M / 16;
  const int groups = ncol / knb;
  const int ntpg   = knb / 16;
  const int tasks  = mt_n * groups;
  for (int t = wave; t < tasks; t += 8) {
    int mt = t % mt_n, g = t / mt_n;
    float vmax[8];
#pragma unroll
    for (int r = 0; r < 8; ++r) vmax[r] = -3.0e38f;
    for (int j = 0; j < ntpg; ++j) {
      v8f c = {};
      for (int kt = 0; kt < kfrags; ++kt) {
        v16h a = load_A_sw(Wsw, kfrags, mt, kt, lane);
        v16h b = load_B_lds(Bsrc, pitchB, kt * 32, (g * ntpg + j) * 16, lane);
        c = wmma32(a, b, c);
      }
      int m0 = mt * 16 + ((lane & 16) ? 8 : 0);
#pragma unroll
      for (int r = 0; r < 8; ++r)
        vmax[r] = fmaxf(vmax[r], c[r] * ss[m0 + r].x + ss[m0 + r].y);
    }
#pragma unroll
    for (int off = 8; off >= 1; off >>= 1)
#pragma unroll
      for (int r = 0; r < 8; ++r)
        vmax[r] = fmaxf(vmax[r], __shfl_xor(vmax[r], off, 32));
    if ((lane & 15) == 0) {
      int m0 = mt * 16 + ((lane & 16) ? 8 : 0);
#pragma unroll
      for (int r = 0; r < 8; ++r)
        outb[(size_t)(m0 + r) * S + sbase + g] = vmax[r];
    }
  }
}

// Gather features into LDS as f16, [col][chan] layout (chan zero-padded):
//   chans 0..2   : grouped_xyz - new_xyz (0 for group-all / sa3)
//   chans 3..C-1 : gathered point features
__device__ inline void stage_feats(half_t* buf, int pitchC, int cinp, int ncol,
                                   const float* xyz_b, int Npts,
                                   const float* fpts_b, int CF,
                                   const float* newxyz_b, int S,
                                   const int* nidx_b, int sbase, int knb,
                                   int tid) {
  for (int i = tid; i < ncol * cinp; i += 256) {
    int col = i / cinp, c = i % cinp;
    int s   = sbase + col / knb;
    int nbr = nidx_b ? nidx_b[(size_t)s * knb + (col % knb)] : col;
    float v;
    if (c < 3)
      v = nidx_b ? (xyz_b[(size_t)c * Npts + nbr] - newxyz_b[(size_t)c * S + s]) : 0.f;
    else if (c < 3 + CF)
      v = fpts_b[(size_t)(c - 3) * Npts + nbr];
    else
      v = 0.f;
    buf[(size_t)col * pitchC + c] = (half_t)v;
  }
}

// ---------------------------------------------------------------------------
// Kernel 0: weight prep — f32 -> f16, swizzled into WMMA A-fragment order.
// dst element i: frag = i>>9, lane = (i>>4)&31, h = i&15
// ---------------------------------------------------------------------------
__global__ __launch_bounds__(256)
void prep_w_kernel(const float* __restrict__ W, int mrows, int K, int kfrags,
                   int total, half_t* __restrict__ dst) {
  int i = blockIdx.x * 256 + threadIdx.x;
  if (i >= total) return;
  int h    = i & 15;
  int lane = (i >> 4) & 31;
  int frag = i >> 9;
  int kt = frag % kfrags, mt = frag / kfrags;
  int m = mt * 16 + (lane & 15);
  int k = kt * 32 + ((lane & 16) ? 16 : 0) + h;
  float v = (m < mrows && k < K) ? W[(size_t)m * K + k] : 0.f;
  dst[i] = (half_t)v;
}

// ---------------------------------------------------------------------------
// Kernel 1: temporal feature MLP, pts(B,8,N)
// ---------------------------------------------------------------------------
__global__ __launch_bounds__(256)
void temporal_kernel(const float* __restrict__ x,
                     const float* te1w, const float* te1b,
                     const float* te2w, const float* te2b,
                     const float* divt,
                     const float* ff1w, const float* ff1b,
                     const float* ff2w, const float* ff2b,
                     float* __restrict__ pts) {
  int i = blockIdx.x * 256 + threadIdx.x;
  if (i >= BATCH * NPTS) return;
  int b = i / NPTS, n = i % NPTS;
  const float* xb = x + (size_t)b * 4 * NPTS;
  float t = xb[2 * NPTS + n], pol = xb[3 * NPTS + n];

  float h1[4];
#pragma unroll
  for (int j = 0; j < 4; ++j) h1[j] = fmaxf(te1w[j] * t + te1b[j], 0.f);

  float comb[9];
#pragma unroll
  for (int j = 0; j < 8; ++j) {
    float tf = te2b[j];
#pragma unroll
    for (int k = 0; k < 4; ++k) tf += te2w[j * 4 + k] * h1[k];
    comb[j] = tf;
  }
#pragma unroll
  for (int k = 0; k < 4; ++k) {
    float ang = t * divt[k];
    comb[2 * k]     += sinf(ang);
    comb[2 * k + 1] += cosf(ang);
  }
  comb[8] = pol;

  float f1[16];
#pragma unroll
  for (int j = 0; j < 16; ++j) {
    float s = ff1b[j];
#pragma unroll
    for (int c = 0; c < 9; ++c) s += ff1w[j * 9 + c] * comb[c];
    f1[j] = fmaxf(s, 0.f);
  }
#pragma unroll
  for (int j = 0; j < 8; ++j) {
    float s = ff2b[j];
#pragma unroll
    for (int c = 0; c < 16; ++c) s += ff2w[j * 16 + c] * f1[c];
    pts[(size_t)b * 8 * NPTS + (size_t)j * NPTS + n] = s;
  }
}

// ---------------------------------------------------------------------------
// Kernel 2: kNN selection. One block per (b,s). Distances cached in LDS once;
// K rounds of incremental argmin (only the winner's owner rescans its chunk).
// Tie-break = lower index, matching lax.top_k.
// ---------------------------------------------------------------------------
__global__ __launch_bounds__(256)
void knn_kernel(const float* __restrict__ xyz, int xyz_bstride, int Npts,
                const int* __restrict__ cidx, const float* __restrict__ cxyz,
                int S, int K, int* __restrict__ nidx,
                float* __restrict__ outcent) {
  extern __shared__ char smem[];
  float* dist = (float*)smem;
  float* tmin = dist + Npts;
  int*   targ = (int*)(tmin + 256);
  float* rvv  = (float*)(targ + 256);
  int*   rii  = (int*)(rvv + 256);

  int tid = threadIdx.x;
  int b = blockIdx.x / S, s = blockIdx.x % S;
  const float* xb = xyz + (size_t)b * xyz_bstride;

  float cx, cy, cz;
  if (cidx) {
    int ci = cidx[s];
    cx = xb[ci]; cy = xb[Npts + ci]; cz = xb[2 * Npts + ci];
  } else {
    const float* cb = cxyz + (size_t)b * 3 * S;
    cx = cb[s]; cy = cb[S + s]; cz = cb[2 * S + s];
  }
  if (outcent && tid == 0) {
    float* ob = outcent + (size_t)b * 3 * S;
    ob[s] = cx; ob[S + s] = cy; ob[2 * S + s] = cz;
  }
  for (int i = tid; i < Npts; i += 256) {
    float dx = xb[i] - cx, dy = xb[Npts + i] - cy, dz = xb[2 * Npts + i] - cz;
    dist[i] = dx * dx + dy * dy + dz * dz;
  }
  __syncthreads();

  const int chunk = Npts / 256;
  const int cbeg = tid * chunk, cend = cbeg + chunk;
  float lm = 3e38f; int la = cbeg;
  for (int j = cbeg; j < cend; ++j) {
    float d = dist[j];
    if (d < lm) { lm = d; la = j; }
  }
  tmin[tid] = lm; targ[tid] = la;
  __syncthreads();

  for (int r = 0; r < K; ++r) {
    rvv[tid] = tmin[tid]; rii[tid] = targ[tid];
    __syncthreads();
    for (int off = 128; off > 0; off >>= 1) {
      if (tid < off) {
        float v2 = rvv[tid + off]; int i2 = rii[tid + off];
        if (v2 < rvv[tid] || (v2 == rvv[tid] && i2 < rii[tid])) {
          rvv[tid] = v2; rii[tid] = i2;
        }
      }
      __syncthreads();
    }
    int widx = rii[0];
    if (tid == 0)
      nidx[(size_t)b * S * K + (size_t)s * K + r] = widx;
    __syncthreads();
    if (widx >= cbeg && widx < cend) {
      dist[widx] = 3e38f;
      float m2 = 3e38f; int a2 = cbeg;
      for (int j = cbeg; j < cend; ++j) {
        float d = dist[j];
        if (d < m2) { m2 = d; a2 = j; }
      }
      tmin[tid] = m2; targ[tid] = a2;
    }
    __syncthreads();
  }
}

// ---------------------------------------------------------------------------
// Kernel 3: fused set-abstraction conv stack (gather -> 3x WMMA GEMM -> max)
// ---------------------------------------------------------------------------
struct ConvArgs {
  const float* xyz;     // (B, 3+, Npts), channel stride Npts
  const float* fpts;    // (B, CF, Npts)
  const float* newxyz;  // (B, 3, S)
  const int*   nidx;    // (B, S, KNB), null => group-all (nbr = col, dxyz = 0)
  const half_t *w1s, *w2s, *w3s;          // swizzled f16 weights
  const float *cb1, *cb2, *cb3;           // conv biases
  const float *g1, *bb1, *rm1, *rv1;
  const float *g2, *bb2, *rm2, *rv2;
  const float *g3, *bb3, *rm3, *rv3;
  float* out;           // (B, C3, S)
  int Npts, S, xyz_bstride, fpts_bstride;
};

template <int CIN, int CINP, int C1, int C2, int C3, int KNB, int TS>
__global__ __launch_bounds__(256)
void sa_conv_kernel(ConvArgs a) {
  constexpr int NCOL = KNB * TS;
  constexpr int PIN  = CINP + 8;
  constexpr int P1   = C1 + 8;
  constexpr int P2   = C2 + 8;
  static_assert(NCOL % 16 == 0 && CINP % 32 == 0 && C1 % 32 == 0 && C2 % 32 == 0, "");

  extern __shared__ char smem[];
  half_t* bufIn = (half_t*)smem;
  half_t* hb1   = bufIn + (size_t)NCOL * PIN;
  half_t* hb2   = hb1 + (size_t)NCOL * P1;
  float2* ss1   = (float2*)(hb2 + (size_t)NCOL * P2);
  float2* ss2   = ss1 + C1;
  float2* ss3   = ss2 + C2;

  int tid = threadIdx.x, lane = tid & 31, wave = tid >> 5;
  int nblkS = a.S / TS;
  int b = blockIdx.x / nblkS;
  int sbase = (blockIdx.x % nblkS) * TS;

  const float* xyz_b  = a.xyz + (size_t)b * a.xyz_bstride;
  const float* fpts_b = a.fpts + (size_t)b * a.fpts_bstride;
  const float* nxyz_b = a.newxyz + (size_t)b * 3 * a.S;
  const int*   nidx_b = a.nidx ? a.nidx + (size_t)b * a.S * KNB : nullptr;
  float*       out_b  = a.out + (size_t)b * C3 * a.S;

  stage_scsh(ss1, a.g1, a.bb1, a.rm1, a.rv1, a.cb1, C1, tid);
  stage_scsh(ss2, a.g2, a.bb2, a.rm2, a.rv2, a.cb2, C2, tid);
  stage_scsh(ss3, a.g3, a.bb3, a.rm3, a.rv3, a.cb3, C3, tid);
  stage_feats(bufIn, PIN, CINP, NCOL, xyz_b, a.Npts, fpts_b, CIN - 3,
              nxyz_b, a.S, nidx_b, sbase, KNB, tid);
  __syncthreads();

  gemm_stage<true>(a.w1s, C1, CINP / 32, bufIn, PIN, hb1, P1, NCOL, ss1, wave, lane);
  __syncthreads();
  gemm_stage<true>(a.w2s, C2, C1 / 32, hb1, P1, hb2, P2, NCOL, ss2, wave, lane);
  __syncthreads();
  gemm3_max(a.w3s, C3, C2 / 32, hb2, P2, ss3, KNB, NCOL, out_b, a.S, sbase,
            wave, lane);
}

// ---------------------------------------------------------------------------
// Kernel 4: farthest point sampling (one block per batch), matches lax.scan:
// emits `far` before updating it; also gathers xyz2.
// ---------------------------------------------------------------------------
__global__ __launch_bounds__(256)
void fps_kernel(const float* __restrict__ xyz1, int* __restrict__ idx2,
                float* __restrict__ xyz2) {
  const int Np = 512, M = 128;
  __shared__ float dist[Np];
  __shared__ float rvv[256];
  __shared__ int   rii[256];
  __shared__ int   sfar;
  int b = blockIdx.x, tid = threadIdx.x;
  const float* xb = xyz1 + (size_t)b * 3 * Np;

  for (int i = tid; i < Np; i += 256) dist[i] = 1e10f;
  if (tid == 0) sfar = 0;
  __syncthreads();

  for (int it = 0; it < M; ++it) {
    int far = sfar;
    if (tid == 0) {
      idx2[(size_t)b * M + it] = far;
      float* ob = xyz2 + (size_t)b * 3 * M;
      ob[it]         = xb[far];
      ob[M + it]     = xb[Np + far];
      ob[2 * M + it] = xb[2 * Np + far];
    }
    float cx = xb[far], cy = xb[Np + far], cz = xb[2 * Np + far];
    float lm = -1.f; int la = 0;
    for (int i = tid; i < Np; i += 256) {
      float dx = xb[i] - cx, dy = xb[Np + i] - cy, dz = xb[2 * Np + i] - cz;
      float nd = fminf(dist[i], dx * dx + dy * dy + dz * dz);
      dist[i] = nd;
      if (nd > lm) { lm = nd; la = i; }
    }
    rvv[tid] = lm; rii[tid] = la;
    __syncthreads();
    for (int off = 128; off > 0; off >>= 1) {
      if (tid < off) {
        float v2 = rvv[tid + off]; int i2 = rii[tid + off];
        if (v2 > rvv[tid] || (v2 == rvv[tid] && i2 < rii[tid])) {
          rvv[tid] = v2; rii[tid] = i2;
        }
      }
      __syncthreads();
    }
    if (tid == 0) sfar = rii[0];
    __syncthreads();
  }
}

// ---------------------------------------------------------------------------
// Kernel 5: FC head. N=16 (batch) = exactly one WMMA N-tile.
// ---------------------------------------------------------------------------
__global__ __launch_bounds__(256)
void head_kernel(const float* __restrict__ feat,
                 const half_t* __restrict__ w1s, const float* fb1,
                 const float* g1, const float* b1, const float* m1, const float* v1,
                 const half_t* __restrict__ w2s, const float* fb2,
                 const float* g2, const float* b2, const float* m2, const float* v2,
                 const half_t* __restrict__ w3s, const float* fb3,
                 float* __restrict__ out) {
  const int P1h = 520;  // 512 + 8
  const int P2h = 264;  // 256 + 8
  __shared__ half_t h1[16 * P1h];
  __shared__ half_t h2[16 * P2h];
  __shared__ float2 ss1[512];
  __shared__ float2 ss2[256];
  int tid = threadIdx.x, lane = tid & 31, wave = tid >> 5;

  stage_scsh(ss1, g1, b1, m1, v1, fb1, 512, tid);
  stage_scsh(ss2, g2, b2, m2, v2, fb2, 256, tid);
  __syncthreads();

  // fc1: 512x1024 @ feat(16x1024)^T
  for (int mt = wave; mt < 32; mt += 8) {
    v8f c = {};
    for (int kt = 0; kt < 32; ++kt) {
      v16h a  = load_A_sw(w1s, 32, mt, kt, lane);
      v16h bf = load_B_gf32(feat, 1024, 1024, kt * 32, lane);
      c = wmma32(a, bf, c);
    }
    int n  = lane & 15;
    int m0 = mt * 16 + ((lane & 16) ? 8 : 0);
    v8h pk;
#pragma unroll
    for (int r = 0; r < 8; ++r)
      pk[r] = (half_t)fmaxf(c[r] * ss1[m0 + r].x + ss1[m0 + r].y, 0.f);
    *(v8h*)&h1[(size_t)n * P1h + m0] = pk;
  }
  __syncthreads();

  // fc2: 256x512
  for (int mt = wave; mt < 16; mt += 8) {
    v8f c = {};
    for (int kt = 0; kt < 16; ++kt) {
      v16h a  = load_A_sw(w2s, 16, mt, kt, lane);
      v16h bf = load_B_lds(h1, P1h, kt * 32, 0, lane);
      c = wmma32(a, bf, c);
    }
    int n  = lane & 15;
    int m0 = mt * 16 + ((lane & 16) ? 8 : 0);
    v8h pk;
#pragma unroll
    for (int r = 0; r < 8; ++r)
      pk[r] = (half_t)fmaxf(c[r] * ss2[m0 + r].x + ss2[m0 + r].y, 0.f);
    *(v8h*)&h2[(size_t)n * P2h + m0] = pk;
  }
  __syncthreads();

  // fc3: 12x256 (M padded to 16)
  if (wave == 0) {
    v8f c = {};
    for (int kt = 0; kt < 8; ++kt) {
      v16h a  = load_A_sw(w3s, 8, 0, kt, lane);
      v16h bf = load_B_lds(h2, P2h, kt * 32, 0, lane);
      c = wmma32(a, bf, c);
    }
    int n = lane & 15;  // batch
#pragma unroll
    for (int r = 0; r < 8; ++r) {
      int o = r + ((lane & 16) ? 8 : 0);
      if (o < NCLS) out[(size_t)n * NCLS + o] = c[r] + fb3[o];
    }
  }
}

// ---------------------------------------------------------------------------
// Host launch. Input order: top-level dict insertion order (x, params, idx1),
// params tree-flattened with sorted keys at every level (jax tree_leaves).
// ---------------------------------------------------------------------------
enum {
  IN_X = 0,
  IN_HBN1_B, IN_HBN1_G, IN_HBN1_RM, IN_HBN1_RV,      // head bn1 (512)
  IN_HBN2_B, IN_HBN2_G, IN_HBN2_RM, IN_HBN2_RV,      // head bn2 (256)
  IN_DIVT,
  IN_FC1_B, IN_FC1_W,
  IN_FC2_B, IN_FC2_W,
  IN_FC3_B, IN_FC3_W,
  IN_FF1_B, IN_FF1_W,
  IN_FF2_B, IN_FF2_W,
  S1_BN1_B, S1_BN1_G, S1_BN1_RM, S1_BN1_RV,
  S1_BN2_B, S1_BN2_G, S1_BN2_RM, S1_BN2_RV,
  S1_BN3_B, S1_BN3_G, S1_BN3_RM, S1_BN3_RV,
  S1_C1_B, S1_C1_W, S1_C2_B, S1_C2_W, S1_C3_B, S1_C3_W,
  S2_BN1_B, S2_BN1_G, S2_BN1_RM, S2_BN1_RV,
  S2_BN2_B, S2_BN2_G, S2_BN2_RM, S2_BN2_RV,
  S2_BN3_B, S2_BN3_G, S2_BN3_RM, S2_BN3_RV,
  S2_C1_B, S2_C1_W, S2_C2_B, S2_C2_W, S2_C3_B, S2_C3_W,
  S3_BN1_B, S3_BN1_G, S3_BN1_RM, S3_BN1_RV,
  S3_BN2_B, S3_BN2_G, S3_BN2_RM, S3_BN2_RV,
  S3_BN3_B, S3_BN3_G, S3_BN3_RM, S3_BN3_RV,
  S3_C1_B, S3_C1_W, S3_C2_B, S3_C2_W, S3_C3_B, S3_C3_W,
  IN_TE1_B, IN_TE1_W, IN_TE2_B, IN_TE2_W,
  IN_IDX1
};

extern "C" void kernel_launch(void* const* d_in, const int* in_sizes, int n_in,
                              void* d_out, int out_size, void* d_ws, size_t ws_size,
                              hipStream_t stream) {
  (void)in_sizes; (void)n_in; (void)out_size; (void)ws_size;
  auto F = [&](int i) { return (const float*)d_in[i]; };
  const float* x    = F(IN_X);
  const int*   idx1 = (const int*)d_in[IN_IDX1];
  float* out = (float*)d_out;

  // workspace carve (256B aligned)
  char* ws = (char*)d_ws;
  size_t off = 0;
  auto carve = [&](size_t bytes) -> void* {
    off = (off + 255) & ~(size_t)255;
    void* p = ws + off;
    off += bytes;
    return p;
  };
  float* pts   = (float*)carve((size_t)BATCH * 8 * NPTS * 4);
  float* xyz1  = (float*)carve((size_t)BATCH * 3 * 512 * 4);
  int*   nidx1 = (int*)carve((size_t)BATCH * 512 * 32 * 4);
  float* p1    = (float*)carve((size_t)BATCH * 128 * 512 * 4);
  int*   idx2  = (int*)carve((size_t)BATCH * 128 * 4);
  float* xyz2  = (float*)carve((size_t)BATCH * 3 * 128 * 4);
  int*   nidx2 = (int*)carve((size_t)BATCH * 128 * 64 * 4);
  float* p2    = (float*)carve((size_t)BATCH * 256 * 128 * 4);
  float* feat3 = (float*)carve((size_t)BATCH * 1024 * 4);

  // weight prep: f32 -> f16 swizzled A-fragments
  auto prep = [&](int widx, int Mpad, int mrows, int K) -> half_t* {
    int kfrags = (K + 31) / 32;
    int total  = Mpad * kfrags * 32;  // halves
    half_t* dst = (half_t*)carve((size_t)total * 2);
    prep_w_kernel<<<(total + 255) / 256, 256, 0, stream>>>(
        F(widx), mrows, K, kfrags, total, dst);
    return dst;
  };
  half_t* s1w1 = prep(S1_C1_W, 64, 64, 11);
  half_t* s1w2 = prep(S1_C2_W, 64, 64, 64);
  half_t* s1w3 = prep(S1_C3_W, 128, 128, 64);
  half_t* s2w1 = prep(S2_C1_W, 128, 128, 131);
  half_t* s2w2 = prep(S2_C2_W, 128, 128, 128);
  half_t* s2w3 = prep(S2_C3_W, 256, 256, 128);
  half_t* s3w1 = prep(S3_C1_W, 256, 256, 259);
  half_t* s3w2 = prep(S3_C2_W, 512, 512, 256);
  half_t* s3w3 = prep(S3_C3_W, 1024, 1024, 512);
  half_t* fw1  = prep(IN_FC1_W, 512, 512, 1024);
  half_t* fw2  = prep(IN_FC2_W, 256, 256, 512);
  half_t* fw3  = prep(IN_FC3_W, 16, 12, 256);

  // dynamic LDS sizes: [col][chan] activation buffers + folded bn params
  const size_t knn1_smem = (size_t)NPTS * 4 + 4096;
  const size_t knn2_smem = (size_t)512 * 4 + 4096;
  auto conv_smem = [](int cinp, int c1, int c2, int c3, int ncol) -> size_t {
    return (size_t)ncol * ((cinp + 8) + (c1 + 8) + (c2 + 8)) * 2 +
           (size_t)(c1 + c2 + c3) * 8;
  };
  const size_t smem1 = conv_smem(32, 64, 64, 128, 256);     // ~96 KB
  const size_t smem2 = conv_smem(160, 128, 128, 256, 128);  // ~117 KB
  const size_t smem3 = conv_smem(288, 256, 512, 1024, 128); // ~291 KB (< 320 KB/WGP)

  (void)hipFuncSetAttribute(reinterpret_cast<const void*>(&knn_kernel),
                            hipFuncAttributeMaxDynamicSharedMemorySize, (int)knn1_smem);
  (void)hipFuncSetAttribute(reinterpret_cast<const void*>(&sa_conv_kernel<11, 32, 64, 64, 128, 32, 8>),
                            hipFuncAttributeMaxDynamicSharedMemorySize, (int)smem1);
  (void)hipFuncSetAttribute(reinterpret_cast<const void*>(&sa_conv_kernel<131, 160, 128, 128, 256, 64, 2>),
                            hipFuncAttributeMaxDynamicSharedMemorySize, (int)smem2);
  (void)hipFuncSetAttribute(reinterpret_cast<const void*>(&sa_conv_kernel<259, 288, 256, 512, 1024, 128, 1>),
                            hipFuncAttributeMaxDynamicSharedMemorySize, (int)smem3);

  // 1) temporal features pts(B,8,N)
  temporal_kernel<<<(BATCH * NPTS) / 256, 256, 0, stream>>>(
      x, F(IN_TE1_W), F(IN_TE1_B), F(IN_TE2_W), F(IN_TE2_B), F(IN_DIVT),
      F(IN_FF1_W), F(IN_FF1_B), F(IN_FF2_W), F(IN_FF2_B), pts);

  // 2) sa1 kNN (centroids = xyz[:, :, idx1]); also writes xyz1
  knn_kernel<<<BATCH * 512, 256, knn1_smem, stream>>>(
      x, 4 * NPTS, NPTS, idx1, nullptr, 512, 32, nidx1, xyz1);

  // 3) sa1 fused conv stack -> p1(B,128,512)
  ConvArgs a1 = {};
  a1.xyz = x;        a1.fpts = pts;  a1.newxyz = xyz1; a1.nidx = nidx1;
  a1.w1s = s1w1; a1.w2s = s1w2; a1.w3s = s1w3;
  a1.cb1 = F(S1_C1_B); a1.cb2 = F(S1_C2_B); a1.cb3 = F(S1_C3_B);
  a1.g1 = F(S1_BN1_G); a1.bb1 = F(S1_BN1_B); a1.rm1 = F(S1_BN1_RM); a1.rv1 = F(S1_BN1_RV);
  a1.g2 = F(S1_BN2_G); a1.bb2 = F(S1_BN2_B); a1.rm2 = F(S1_BN2_RM); a1.rv2 = F(S1_BN2_RV);
  a1.g3 = F(S1_BN3_G); a1.bb3 = F(S1_BN3_B); a1.rm3 = F(S1_BN3_RM); a1.rv3 = F(S1_BN3_RV);
  a1.out = p1; a1.Npts = NPTS; a1.S = 512;
  a1.xyz_bstride = 4 * NPTS; a1.fpts_bstride = 8 * NPTS;
  sa_conv_kernel<11, 32, 64, 64, 128, 32, 8>
      <<<BATCH * (512 / 8), 256, smem1, stream>>>(a1);

  // 4) FPS on xyz1 -> idx2, xyz2
  fps_kernel<<<BATCH, 256, 0, stream>>>(xyz1, idx2, xyz2);

  // 5) sa2 kNN (centroids = xyz2, neighbors among xyz1)
  knn_kernel<<<BATCH * 128, 256, knn2_smem, stream>>>(
      xyz1, 3 * 512, 512, nullptr, xyz2, 128, 64, nidx2, nullptr);

  // 6) sa2 fused conv stack -> p2(B,256,128)
  ConvArgs a2 = {};
  a2.xyz = xyz1;     a2.fpts = p1;   a2.newxyz = xyz2; a2.nidx = nidx2;
  a2.w1s = s2w1; a2.w2s = s2w2; a2.w3s = s2w3;
  a2.cb1 = F(S2_C1_B); a2.cb2 = F(S2_C2_B); a2.cb3 = F(S2_C3_B);
  a2.g1 = F(S2_BN1_G); a2.bb1 = F(S2_BN1_B); a2.rm1 = F(S2_BN1_RM); a2.rv1 = F(S2_BN1_RV);
  a2.g2 = F(S2_BN2_G); a2.bb2 = F(S2_BN2_B); a2.rm2 = F(S2_BN2_RM); a2.rv2 = F(S2_BN2_RV);
  a2.g3 = F(S2_BN3_G); a2.bb3 = F(S2_BN3_B); a2.rm3 = F(S2_BN3_RM); a2.rv3 = F(S2_BN3_RV);
  a2.out = p2; a2.Npts = 512; a2.S = 128;
  a2.xyz_bstride = 3 * 512; a2.fpts_bstride = 128 * 512;
  sa_conv_kernel<131, 160, 128, 128, 256, 64, 2>
      <<<BATCH * (128 / 2), 256, smem2, stream>>>(a2);

  // 7) sa3 group-all fused stack + global max -> feat3(B,1024)
  ConvArgs a3 = {};
  a3.xyz = xyz2;     a3.fpts = p2;   a3.newxyz = xyz2; a3.nidx = nullptr;
  a3.w1s = s3w1; a3.w2s = s3w2; a3.w3s = s3w3;
  a3.cb1 = F(S3_C1_B); a3.cb2 = F(S3_C2_B); a3.cb3 = F(S3_C3_B);
  a3.g1 = F(S3_BN1_G); a3.bb1 = F(S3_BN1_B); a3.rm1 = F(S3_BN1_RM); a3.rv1 = F(S3_BN1_RV);
  a3.g2 = F(S3_BN2_G); a3.bb2 = F(S3_BN2_B); a3.rm2 = F(S3_BN2_RM); a3.rv2 = F(S3_BN2_RV);
  a3.g3 = F(S3_BN3_G); a3.bb3 = F(S3_BN3_B); a3.rm3 = F(S3_BN3_RM); a3.rv3 = F(S3_BN3_RV);
  a3.out = feat3; a3.Npts = 128; a3.S = 1;
  a3.xyz_bstride = 3 * 128; a3.fpts_bstride = 256 * 128;
  sa_conv_kernel<259, 288, 256, 512, 1024, 128, 1>
      <<<BATCH, 256, smem3, stream>>>(a3);

  // 8) FC head -> out(16,12)
  head_kernel<<<1, 256, 0, stream>>>(
      feat3,
      fw1, F(IN_FC1_B),
      F(IN_HBN1_G), F(IN_HBN1_B), F(IN_HBN1_RM), F(IN_HBN1_RV),
      fw2, F(IN_FC2_B),
      F(IN_HBN2_G), F(IN_HBN2_B), F(IN_HBN2_RM), F(IN_HBN2_RV),
      fw3, F(IN_FC3_B),
      out);
}
